// FusionModel_45904610460051
// MI455X (gfx1250) — matrix operand order repaired and checked by reading, not compiled, not run on hardware
//
#include <hip/hip_runtime.h>

#define DD    1536
#define HH    12
#define HDIM  128
#define DHALF 64
#define FFND  8960
#define SVLEN 2048
#define SALEN 1024
#define LCTX  512

typedef unsigned short u16t;
typedef unsigned int   u32t;
typedef __attribute__((ext_vector_type(16))) __bf16 v16bf;
typedef __attribute__((ext_vector_type(8)))  float  v8f;

__device__ __forceinline__ u16t f2bf(float f) {
  u32t u = __float_as_uint(f);
  u32t r = (u + 0x7FFFu + ((u >> 16) & 1u)) >> 16;   // round-to-nearest-even
  return (u16t)r;
}
__device__ __forceinline__ u32t pk2(float a, float b) {
  return (u32t)f2bf(a) | ((u32t)f2bf(b) << 16);
}
// branch-free tanh via hardware v_exp_f32 (avoids libm's divergent path)
__device__ __forceinline__ float fast_tanh(float u) {
  float t = __expf(-2.f * __builtin_fabsf(u));
  float th = (1.f - t) / (1.f + t);
  return __builtin_copysignf(th, u);
}

union FragU { struct { uint4 lo, hi; } u; v16bf v; };

// A-matrix fragment (16x32 bf16): lane L<16 holds row L, K 0..7 (+0B) and K 16..23 (+32B);
// lane L+16 holds K 8..15 (+16B) and K 24..31 (+48B).  strideH in u16 units.
__device__ __forceinline__ v16bf fragA(const u16t* base, int strideH, int lane) {
  int r = lane & 15, h = lane >> 4;
  const u16t* p = base + r * strideH + h * 8;
  FragU t;
  t.u.lo = *(const uint4*)p;
  t.u.hi = *(const uint4*)(p + 16);
  return t.v;
}
// B-matrix fragment (32x16 bf16): lane n<16 holds column n, K 0..7 and K 16..23; lane n+16 holds K 8..15 and K 24..31.
__device__ __forceinline__ v16bf fragB(const u16t* base, int strideH, int lane) {
  int r = lane & 15, h = lane >> 4;
  const u16t* p = base + r * strideH + h * 8;
  FragU t;
  t.u.lo = *(const uint4*)p;
  t.u.hi = *(const uint4*)(p + 16);
  return t.v;
}
__device__ __forceinline__ v8f wmma_bf(v16bf a, v16bf b, v8f c) {
  return __builtin_amdgcn_wmma_f32_16x16x32_bf16(false, a, false, b, (short)0, c, false, false);
}

// ---------------- weight / activation conversion ----------------

// X f32 [K][N]  ->  Y bf16 [N][K]   (K,N multiples of 32)
__global__ __launch_bounds__(256) void k_tr_bf16(const float* __restrict__ X,
                                                 u16t* __restrict__ Y, int K, int N) {
  __shared__ float tile[32][33];
  int t = threadIdx.x, tx = t & 31, ty = t >> 5;
  int k0 = blockIdx.y * 32, n0 = blockIdx.x * 32;
  for (int r = ty; r < 32; r += 8)
    tile[r][tx] = X[(size_t)(k0 + r) * N + n0 + tx];
  __syncthreads();
  for (int r = ty; r < 32; r += 8)
    Y[(size_t)(n0 + r) * K + k0 + tx] = f2bf(tile[tx][r]);
}

__global__ void k_to_bf16(const float* __restrict__ X, u16t* __restrict__ Y, int n) {
  int i = blockIdx.x * blockDim.x + threadIdx.x;
  if ((i << 2) >= n) return;
  float4 f = ((const float4*)X)[i];
  ((uint2*)Y)[i] = make_uint2(pk2(f.x, f.y), pk2(f.z, f.w));
}

// ---------------- layer norms ----------------

// y = ln(x); if g: y = y*g + b; if m1: y = y*(1 + m1 + e1) + (m0 + e0)
__global__ __launch_bounds__(256) void k_ln(const float* __restrict__ x, float* __restrict__ y,
                                            const float* __restrict__ g, const float* __restrict__ b,
                                            const float* __restrict__ m0, const float* __restrict__ e0,
                                            const float* __restrict__ m1, const float* __restrict__ e1) {
  __shared__ float red[256];
  int s = blockIdx.x, t = threadIdx.x;
  const float* xr = x + (size_t)s * DD;
  float v[6], sum = 0.f;
#pragma unroll
  for (int i = 0; i < 6; i++) { v[i] = xr[t + i * 256]; sum += v[i]; }
  red[t] = sum; __syncthreads();
  for (int o = 128; o > 0; o >>= 1) { if (t < o) red[t] += red[t + o]; __syncthreads(); }
  float mean = red[0] * (1.f / DD);
  __syncthreads();
  float sq = 0.f;
#pragma unroll
  for (int i = 0; i < 6; i++) { float d = v[i] - mean; sq += d * d; }
  red[t] = sq; __syncthreads();
  for (int o = 128; o > 0; o >>= 1) { if (t < o) red[t] += red[t + o]; __syncthreads(); }
  float rstd = rsqrtf(red[0] * (1.f / DD) + 1e-6f);
  float* yr = y + (size_t)s * DD;
#pragma unroll
  for (int i = 0; i < 6; i++) {
    int c = t + i * 256;
    float o = (v[i] - mean) * rstd;
    if (g)  o = o * g[c] + b[c];
    if (m1) o = o * (1.f + m1[c] + e1[c]) + (m0[c] + e0[c]);
    yr[c] = o;
  }
}

__global__ __launch_bounds__(256) void k_rms(const float* __restrict__ x, float* __restrict__ y,
                                             const float* __restrict__ g) {
  __shared__ float red[256];
  int s = blockIdx.x, t = threadIdx.x;
  const float* xr = x + (size_t)s * DD;
  float v[6], sq = 0.f;
#pragma unroll
  for (int i = 0; i < 6; i++) { v[i] = xr[t + i * 256]; sq += v[i] * v[i]; }
  red[t] = sq; __syncthreads();
  for (int o = 128; o > 0; o >>= 1) { if (t < o) red[t] += red[t + o]; __syncthreads(); }
  float rr = rsqrtf(red[0] * (1.f / DD) + 1e-6f);
  float* yr = y + (size_t)s * DD;
#pragma unroll
  for (int i = 0; i < 6; i++) { int c = t + i * 256; yr[c] = v[i] * rr * g[c]; }
}

__global__ void k_rope(const float* __restrict__ x, float* __restrict__ y,
                       const float* __restrict__ cs, const float* __restrict__ sn, int S) {
  int i = blockIdx.x * blockDim.x + threadIdx.x;
  if (i >= S * HH * DHALF) return;
  int s = i / (HH * DHALF);
  int r = i - s * (HH * DHALF);
  int h = r / DHALF, d = r - h * DHALF;
  float c = cs[s * DHALF + d], si = sn[s * DHALF + d];
  size_t base = (size_t)s * DD + h * HDIM + 2 * d;
  float xr = x[base], xi = x[base + 1];
  y[base]     = xr * c - xi * si;
  y[base + 1] = xr * si + xi * c;
}

// out = xin + yv * (m + e)   (m==null -> gate 1)
__global__ void k_resgate(const float* __restrict__ xin, const float* __restrict__ yv,
                          float* __restrict__ out, const float* __restrict__ m,
                          const float* __restrict__ e, int S) {
  int i = blockIdx.x * blockDim.x + threadIdx.x;
  if (i >= S * DD) return;
  int c = i % DD;
  float gate = m ? (m[c] + e[c]) : 1.f;
  out[i] = xin[i] + yv[i] * gate;
}

// ---------------- WMMA GEMM:  C[M][N] = A[M][K] * Wt^T + bias, Wt is bf16 [N][K] ----------------
// Software pipelined: global loads for tile i+1 issue while tile i's WMMAs run.

__global__ __launch_bounds__(256) void k_gemm(const float* __restrict__ A,
                                              const u16t* __restrict__ Wt,
                                              const float* __restrict__ bias,
                                              float* __restrict__ C,
                                              int M, int N, int K, int act) {
  __shared__ u16t As[128 * 40];   // 128 rows x 32 bf16 + pad, 80B stride
  __shared__ u16t Bs[128 * 40];
  int tid = threadIdx.x, lane = tid & 31, wid = tid >> 5;
  int bm = blockIdx.y * 128, bn = blockIdx.x * 128;
  int wm = (wid >> 1) * 32, wn = (wid & 1) * 64;
  v8f acc[2][4] = {};
  int row = tid >> 1, half = tid & 1;
  const float* ap = A + (size_t)(bm + row) * K + half * 16;
  const u16t*  bp = Wt + (size_t)(bn + row) * K + half * 16;
  u16t* asd = As + row * 40 + half * 16;
  u16t* bsd = Bs + row * 40 + half * 16;

  // prologue: stage tile 0 into registers
  float4 f0 = *(const float4*)(ap);
  float4 f1 = *(const float4*)(ap + 4);
  float4 f2 = *(const float4*)(ap + 8);
  float4 f3 = *(const float4*)(ap + 12);
  uint4  wb0 = *(const uint4*)(bp);
  uint4  wb1 = *(const uint4*)(bp + 8);

  for (int kt = 0; kt < K; kt += 32) {
    *(uint4*)asd       = make_uint4(pk2(f0.x, f0.y), pk2(f0.z, f0.w), pk2(f1.x, f1.y), pk2(f1.z, f1.w));
    *(uint4*)(asd + 8) = make_uint4(pk2(f2.x, f2.y), pk2(f2.z, f2.w), pk2(f3.x, f3.y), pk2(f3.z, f3.w));
    *(uint4*)bsd       = wb0;
    *(uint4*)(bsd + 8) = wb1;
    __syncthreads();

    // issue next tile's global loads (overlap with WMMA below); uniform select avoids branches
    int ktn = (kt + 32 < K) ? (kt + 32) : 0;
    f0 = *(const float4*)(ap + ktn);
    f1 = *(const float4*)(ap + ktn + 4);
    f2 = *(const float4*)(ap + ktn + 8);
    f3 = *(const float4*)(ap + ktn + 12);
    wb0 = *(const uint4*)(bp + ktn);
    wb1 = *(const uint4*)(bp + ktn + 8);

    v16bf a0 = fragA(As + (wm + 0)  * 40, 40, lane);
    v16bf a1 = fragA(As + (wm + 16) * 40, 40, lane);
#pragma unroll
    for (int j = 0; j < 4; j++) {
      v16bf bf = fragB(Bs + (wn + j * 16) * 40, 40, lane);
      acc[0][j] = wmma_bf(a0, bf, acc[0][j]);
      acc[1][j] = wmma_bf(a1, bf, acc[1][j]);
    }
    __syncthreads();
  }

  int r = lane & 15, hi = lane >> 4;
#pragma unroll
  for (int i = 0; i < 2; i++)
#pragma unroll
    for (int j = 0; j < 4; j++) {
      int gcol = bn + wn + j * 16 + r;
      float bb = bias[gcol];                     // bias always present in this model
#pragma unroll
      for (int v = 0; v < 8; v++) {
        int grow = bm + wm + i * 16 + hi * 8 + v;
        float val = acc[i][j][v] + bb;
        if (act == 1)
          val = 0.5f * val * (1.f + fast_tanh(0.7978845608f * (val + 0.044715f * val * val * val)));
        C[(size_t)grow * N + gcol] = val;
      }
    }
}

// ---------------- flash attention (WMMA, online softmax, pipelined) ----------------
// Qb,Kb: bf16 [S][DD]; Vt: bf16 [DD][Sk]; Out f32 [Sq][DD]. grid = (Sq/64, H), block = 128.
__global__ __launch_bounds__(128) void k_attn(const u16t* __restrict__ Qb,
                                              const u16t* __restrict__ Kb,
                                              const u16t* __restrict__ Vt,
                                              float* __restrict__ Out,
                                              int Sq, int Sk, int accum) {
  __shared__ u16t Ks[32 * 136];     // 32 keys x 128 hd, 272B stride
  __shared__ u16t Vs[128 * 40];     // 128 hd x 32 keys, 80B stride
  __shared__ u16t Ps[4 * 16 * 40];  // per-wave 16q x 32key P tile (wave-private)
  int tid = threadIdx.x, lane = tid & 31, wid = tid >> 5;
  int h = blockIdx.y;
  int q0 = blockIdx.x * 64 + wid * 16;
  int r = lane & 15, hi = lane >> 4;

  v16bf q[4];
#pragma unroll
  for (int f = 0; f < 4; f++)
    q[f] = fragA(Qb + (size_t)q0 * DD + h * HDIM + f * 32, DD, lane);

  v8f o[8] = {};
  float mrow[8], lrow[8];
#pragma unroll
  for (int v = 0; v < 8; v++) { mrow[v] = -1e30f; lrow[v] = 0.f; }
  const float scale = 0.08838834764831845f;  // 1/sqrt(128)
  u16t* Pw = Ps + wid * 16 * 40;

  int krow = tid >> 2, sub = tid & 3;
  const u16t* ksrc = Kb + (size_t)krow * DD + h * HDIM + sub * 32;
  const u16t* vsrc = Vt + (size_t)(h * HDIM + tid) * Sk;
  u16t* kdst = Ks + krow * 136 + sub * 32;
  u16t* vdst = Vs + tid * 40;

  // prologue: stage chunk 0 into registers
  uint4 kr0 = *(const uint4*)(ksrc);
  uint4 kr1 = *(const uint4*)(ksrc + 8);
  uint4 kr2 = *(const uint4*)(ksrc + 16);
  uint4 kr3 = *(const uint4*)(ksrc + 24);
  uint4 vr0 = *(const uint4*)(vsrc);
  uint4 vr1 = *(const uint4*)(vsrc + 8);
  uint4 vr2 = *(const uint4*)(vsrc + 16);
  uint4 vr3 = *(const uint4*)(vsrc + 24);

  for (int k0 = 0; k0 < Sk; k0 += 32) {
    __syncthreads();             // all waves finished reading previous Ks/Vs
    *(uint4*)kdst        = kr0;
    *(uint4*)(kdst + 8)  = kr1;
    *(uint4*)(kdst + 16) = kr2;
    *(uint4*)(kdst + 24) = kr3;
    *(uint4*)vdst        = vr0;
    *(uint4*)(vdst + 8)  = vr1;
    *(uint4*)(vdst + 16) = vr2;
    *(uint4*)(vdst + 24) = vr3;
    __syncthreads();

    // issue next chunk's global loads; overlaps with QK^T/softmax/PV below
    int kn = (k0 + 32 < Sk) ? (k0 + 32) : 0;
    {
      const u16t* ks = ksrc + (size_t)kn * DD;
      kr0 = *(const uint4*)(ks);
      kr1 = *(const uint4*)(ks + 8);
      kr2 = *(const uint4*)(ks + 16);
      kr3 = *(const uint4*)(ks + 24);
      const u16t* vs = vsrc + kn;
      vr0 = *(const uint4*)(vs);
      vr1 = *(const uint4*)(vs + 8);
      vr2 = *(const uint4*)(vs + 16);
      vr3 = *(const uint4*)(vs + 24);
    }

    v8f s0 = {}, s1 = {};
#pragma unroll
    for (int f = 0; f < 4; f++) {
      v16bf kb0 = fragB(Ks + 0  * 136 + f * 32, 136, lane);
      v16bf kb1 = fragB(Ks + 16 * 136 + f * 32, 136, lane);
      s0 = wmma_bf(q[f], kb0, s0);
      s1 = wmma_bf(q[f], kb1, s1);
    }

#pragma unroll
    for (int v = 0; v < 8; v++) {
      float a = s0[v] * scale, b = s1[v] * scale;
      float mx = fmaxf(a, b);
#pragma unroll
      for (int m = 8; m >= 1; m >>= 1) mx = fmaxf(mx, __shfl_xor(mx, m, 32));
      float nm = fmaxf(mrow[v], mx);
      float alpha = __expf(mrow[v] - nm);
      float p0 = __expf(a - nm), p1 = __expf(b - nm);
      float rs = p0 + p1;
#pragma unroll
      for (int m = 8; m >= 1; m >>= 1) rs += __shfl_xor(rs, m, 32);
      lrow[v] = lrow[v] * alpha + rs;
      mrow[v] = nm;
#pragma unroll
      for (int j = 0; j < 8; j++) o[j][v] *= alpha;
      int prow = v + hi * 8;
      Pw[prow * 40 + r]      = f2bf(p0);
      Pw[prow * 40 + 16 + r] = f2bf(p1);
    }
    // No block barrier needed: Pw is wave-private and same-wave LDS ops are in-order (DScnt).
    v16bf pa = fragA(Pw, 40, lane);
#pragma unroll
    for (int j = 0; j < 8; j++) {
      v16bf vb = fragB(Vs + (j * 16) * 40, 40, lane);
      o[j] = wmma_bf(pa, vb, o[j]);
    }
  }

#pragma unroll
  for (int j = 0; j < 8; j++)
#pragma unroll
    for (int v = 0; v < 8; v++) {
      int grow = q0 + v + hi * 8;
      int gcol = h * HDIM + j * 16 + r;
      size_t idx = (size_t)grow * DD + gcol;
      float val = o[j][v] / lrow[v];
      if (accum) val += Out[idx];
      Out[idx] = val;
    }
}

// ---------------- host orchestration ----------------

enum { P_MOD = 0, P_SQW, P_SQB, P_SKW, P_SKB, P_SVW, P_SVB, P_SOW, P_SOB,
       P_CQW, P_CQB, P_CKW, P_CKB, P_CVW, P_CVB, P_COW, P_COB,
       P_KFW, P_KFB, P_VFW, P_VFB, P_F1W, P_F1B, P_F2W, P_F2B,
       P_SQG, P_SKG, P_CQG, P_CKG, P_KFG, P_N3G, P_PFG, P_N3B, P_PFB };

struct WB { u16t *sq, *sk, *sv, *so, *cq, *ck, *cv, *co, *kf, *vf, *f1, *f2; };
struct SB {
  float *xm, *ta, *tb, *ao, *yo, *x1, *x2, *xout;
  u16t *qb, *kb, *vtb;
  const float *x0, *e, *ctx, *cs, *sn;
  int S, base;
};

extern "C" void kernel_launch(void* const* d_in, const int* in_sizes, int n_in,
                              void* d_out, int out_size, void* d_ws, size_t ws_size,
                              hipStream_t stream) {
  (void)in_sizes; (void)n_in; (void)out_size; (void)ws_size;
  auto pp = [&](int base, int i) -> const float* { return (const float*)d_in[base + i]; };

  char* ws = (char*)d_ws;
  size_t off = 0;
  auto alloc = [&](size_t bytes) -> void* {
    void* p = ws + off;
    off = (off + bytes + 255) & ~(size_t)255;
    return p;
  };

  auto TR   = [&](const float* X, u16t* Y, int K, int N) {
    k_tr_bf16<<<dim3(N / 32, K / 32), 256, 0, stream>>>(X, Y, K, N);
  };
  auto TOBF = [&](const float* X, u16t* Y, size_t n) {
    k_to_bf16<<<(unsigned)(n / 4 / 256), 256, 0, stream>>>(X, Y, (int)n);
  };
  auto LN   = [&](const float* x, float* y, const float* g, const float* b,
                  const float* m0, const float* e0, const float* m1, const float* e1, int S) {
    k_ln<<<S, 256, 0, stream>>>(x, y, g, b, m0, e0, m1, e1);
  };
  auto RMS  = [&](const float* x, float* y, const float* g, int S) {
    k_rms<<<S, 256, 0, stream>>>(x, y, g);
  };
  auto ROPE = [&](const float* x, float* y, const float* c, const float* s, int S) {
    k_rope<<<(S * HH * DHALF) / 256, 256, 0, stream>>>(x, y, c, s, S);
  };
  auto GEMM = [&](const float* A, const u16t* Wt, const float* bias, float* C,
                  int M, int N, int K, int act) {
    k_gemm<<<dim3(N / 128, M / 128), 256, 0, stream>>>(A, Wt, bias, C, M, N, K, act);
  };
  auto ATTN = [&](const u16t* Qb, const u16t* Kb, const u16t* Vtb, float* Ot,
                  int Sq, int Sk, int acc) {
    k_attn<<<dim3(Sq / 64, HH), 128, 0, stream>>>(Qb, Kb, Vtb, Ot, Sq, Sk, acc);
  };
  auto GATE = [&](const float* xin, const float* yv, float* out,
                  const float* m, const float* e, int S) {
    k_resgate<<<(S * DD) / 256, 256, 0, stream>>>(xin, yv, out, m, e, S);
  };

  // --- pre-transpose all weights to bf16 [N][K] (conflict-free WMMA B staging) ---
  auto prep = [&](int base) -> WB {
    WB w;
    const size_t dd = (size_t)DD * DD * 2;
    w.sq = (u16t*)alloc(dd); TR(pp(base, P_SQW), w.sq, DD, DD);
    w.sk = (u16t*)alloc(dd); TR(pp(base, P_SKW), w.sk, DD, DD);
    w.sv = (u16t*)alloc(dd); TR(pp(base, P_SVW), w.sv, DD, DD);
    w.so = (u16t*)alloc(dd); TR(pp(base, P_SOW), w.so, DD, DD);
    w.cq = (u16t*)alloc(dd); TR(pp(base, P_CQW), w.cq, DD, DD);
    w.ck = (u16t*)alloc(dd); TR(pp(base, P_CKW), w.ck, DD, DD);
    w.cv = (u16t*)alloc(dd); TR(pp(base, P_CVW), w.cv, DD, DD);
    w.co = (u16t*)alloc(dd); TR(pp(base, P_COW), w.co, DD, DD);
    w.kf = (u16t*)alloc(dd); TR(pp(base, P_KFW), w.kf, DD, DD);
    w.vf = (u16t*)alloc(dd); TR(pp(base, P_VFW), w.vf, DD, DD);
    w.f1 = (u16t*)alloc((size_t)DD * FFND * 2); TR(pp(base, P_F1W), w.f1, DD, FFND);
    w.f2 = (u16t*)alloc((size_t)DD * FFND * 2); TR(pp(base, P_F2W), w.f2, FFND, DD);
    return w;
  };
  WB wv = prep(10);
  WB wa = prep(44);

  auto mkbufs = [&](int S) {
    SB s{};
    size_t fb = (size_t)S * DD * 4, hb = (size_t)S * DD * 2;
    s.xm = (float*)alloc(fb); s.ta = (float*)alloc(fb); s.tb = (float*)alloc(fb);
    s.ao = (float*)alloc(fb); s.yo = (float*)alloc(fb);
    s.x1 = (float*)alloc(fb); s.x2 = (float*)alloc(fb);
    s.qb = (u16t*)alloc(hb); s.kb = (u16t*)alloc(hb); s.vtb = (u16t*)alloc(hb);
    s.S = S;
    return s;
  };
  SB sv = mkbufs(SVLEN);
  SB sa = mkbufs(SALEN);
  sv.x0 = (const float*)d_in[0]; sa.x0 = (const float*)d_in[1];
  sv.e  = (const float*)d_in[2]; sa.e  = (const float*)d_in[3];
  sv.ctx = (const float*)d_in[4]; sa.ctx = (const float*)d_in[5];
  sv.cs = (const float*)d_in[6]; sv.sn = (const float*)d_in[7];
  sa.cs = (const float*)d_in[8]; sa.sn = (const float*)d_in[9];
  sv.base = 10; sa.base = 44;
  sv.xout = (float*)d_out;
  sa.xout = (float*)d_out + (size_t)SVLEN * DD;

  float* tgtln = (float*)alloc((size_t)SVLEN * DD * 4);
  float* tgta  = (float*)alloc((size_t)SVLEN * DD * 4);
  float* tgtb  = (float*)alloc((size_t)SVLEN * DD * 4);
  u16t*  ktb   = (u16t*)alloc((size_t)SVLEN * DD * 2);
  u16t*  vt2b  = (u16t*)alloc((size_t)SVLEN * DD * 2);
  float* hid   = (float*)alloc((size_t)SVLEN * FFND * 4);

  auto self_attn = [&](SB& s, const WB& w) {
    const float* mod = pp(s.base, P_MOD);
    LN(s.x0, s.xm, nullptr, nullptr, mod, s.e, mod + DD, s.e + DD, s.S);
    GEMM(s.xm, w.sq, pp(s.base, P_SQB), s.ta, s.S, DD, DD, 0);
    RMS(s.ta, s.tb, pp(s.base, P_SQG), s.S);
    ROPE(s.tb, s.ta, s.cs, s.sn, s.S);
    TOBF(s.ta, s.qb, (size_t)s.S * DD);
    GEMM(s.xm, w.sk, pp(s.base, P_SKB), s.ta, s.S, DD, DD, 0);
    RMS(s.ta, s.tb, pp(s.base, P_SKG), s.S);
    ROPE(s.tb, s.ta, s.cs, s.sn, s.S);
    TOBF(s.ta, s.kb, (size_t)s.S * DD);
    GEMM(s.xm, w.sv, pp(s.base, P_SVB), s.ta, s.S, DD, DD, 0);
    TR(s.ta, s.vtb, s.S, DD);
    ATTN(s.qb, s.kb, s.vtb, s.ao, s.S, s.S, 0);
    GEMM(s.ao, w.so, pp(s.base, P_SOB), s.yo, s.S, DD, DD, 0);
    GATE(s.x0, s.yo, s.x1, mod + 2 * DD, s.e + 2 * DD, s.S);
  };

  auto cross_ffn = [&](SB& s, const WB& w, const SB& t, const float* tcs, const float* tsn) {
    const float* mod = pp(s.base, P_MOD);
    LN(s.x1, s.xm, pp(s.base, P_N3G), pp(s.base, P_N3B), nullptr, nullptr, nullptr, nullptr, s.S);
    GEMM(s.xm, w.cq, pp(s.base, P_CQB), s.ta, s.S, DD, DD, 0);
    RMS(s.ta, s.tb, pp(s.base, P_CQG), s.S);                 // tb = q (kept for both attns)
    GEMM(s.ctx, w.ck, pp(s.base, P_CKB), s.ao, LCTX, DD, DD, 0);
    RMS(s.ao, s.yo, pp(s.base, P_CKG), LCTX);
    TOBF(s.yo, s.kb, (size_t)LCTX * DD);
    GEMM(s.ctx, w.cv, pp(s.base, P_CVB), s.ao, LCTX, DD, DD, 0);
    TR(s.ao, s.vtb, LCTX, DD);
    TOBF(s.tb, s.qb, (size_t)s.S * DD);
    ATTN(s.qb, s.kb, s.vtb, s.ao, s.S, LCTX, 0);
    // target branch
    LN(t.x1, tgtln, pp(s.base, P_PFG), pp(s.base, P_PFB), nullptr, nullptr, nullptr, nullptr, t.S);
    GEMM(tgtln, w.kf, pp(s.base, P_KFB), tgta, t.S, DD, DD, 0);
    RMS(tgta, tgtb, pp(s.base, P_KFG), t.S);
    ROPE(tgtb, tgta, tcs, tsn, t.S);
    TOBF(tgta, ktb, (size_t)t.S * DD);
    GEMM(tgtln, w.vf, pp(s.base, P_VFB), tgta, t.S, DD, DD, 0);
    TR(tgta, vt2b, t.S, DD);
    ROPE(s.tb, s.ta, s.cs, s.sn, s.S);                        // qr = rope(q)
    TOBF(s.ta, s.qb, (size_t)s.S * DD);
    ATTN(s.qb, ktb, vt2b, s.ao, s.S, t.S, 1);                 // accumulate into first attn
    GEMM(s.ao, w.co, pp(s.base, P_COB), s.yo, s.S, DD, DD, 0);
    GATE(s.x1, s.yo, s.x2, nullptr, nullptr, s.S);
    // FFN
    LN(s.x2, s.xm, nullptr, nullptr, mod + 3 * DD, s.e + 3 * DD, mod + 4 * DD, s.e + 4 * DD, s.S);
    GEMM(s.xm, w.f1, pp(s.base, P_F1B), hid, s.S, FFND, DD, 1);
    GEMM(hid, w.f2, pp(s.base, P_F2B), s.yo, s.S, DD, FFND, 0);
    GATE(s.x2, s.yo, s.xout, mod + 5 * DD, s.e + 5 * DD, s.S);
  };

  self_attn(sa, wa);
  self_attn(sv, wv);
  cross_ffn(sa, wa, sv, sv.cs, sv.sn);   // audio cross: target = vid (post-self)
  cross_ffn(sv, wv, sa, sa.cs, sa.sn);   // vid cross: target = og audio (post-self x1)
}
